// Layer_Experts_16260746182871
// MI455X (gfx1250) — matrix operand order repaired and checked by reading, not compiled, not run on hardware
//
#include <hip/hip_runtime.h>
#include <hip/hip_bf16.h>
#include <math.h>

#define NB 4
#define NN 4096
#define DD 1024
#define FF 4096
#define LN_EPS 1e-5f

typedef __attribute__((ext_vector_type(16))) __bf16 v16bf;
typedef __attribute__((ext_vector_type(8)))  __bf16 v8bf;
typedef __attribute__((ext_vector_type(8)))  float  v8f;

__device__ __forceinline__ __bf16 f2bf(float x) {
  union { float f; unsigned u; } c; c.f = x;
  unsigned r = (c.u + 0x7FFFu + ((c.u >> 16) & 1u)) >> 16;   // RNE
  union { unsigned short s; __bf16 b; } o; o.s = (unsigned short)r;
  return o.b;
}

// A-fragment (16x32 bf16, row-major source): lane<16 -> row=lane, K 0..15;
// lane>=16 -> same rows, K 16..31. Caller passes p already offset by kh16.
__device__ __forceinline__ v16bf load_a16(const __bf16* p) {
  v8bf lo = *(const v8bf*)p;
  v8bf hi = *(const v8bf*)(p + 8);
  v16bf r;
#pragma unroll
  for (int i = 0; i < 8; ++i) { r[i] = lo[i]; r[i + 8] = hi[i]; }
  return r;
}

__device__ __forceinline__ float gelu_exact(float v) {
  return 0.5f * v * (1.0f + erff(v * 0.70710678118654752f));
}

// ---------------- prep: fp32 -> bf16 elementwise ----------------
__global__ void k_cvt_bf16(const float* __restrict__ src, __bf16* __restrict__ dst, int n) {
  int i = blockIdx.x * blockDim.x + threadIdx.x;
  if (i < n) dst[i] = f2bf(src[i]);
}

// ------- prep: xT[b][d][n] = bf16(x[b][n][d]) via async global->LDS DMA -------
// 256 threads; each thread DMAs one 16B chunk of the 32x32 fp32 tile into LDS
// (global_load_async_to_lds_b128, tracked by ASYNCcnt), then writes transposed bf16.
__global__ void k_transpose(const float* __restrict__ x, __bf16* __restrict__ xT) {
  __shared__ float tile[32][36];                       // pitch 144B: 16B-aligned chunks
  int b  = blockIdx.z;
  int n0 = blockIdx.x * 32, d0 = blockIdx.y * 32;
  int t  = threadIdx.x;                                // 0..255
  int row = t >> 3;                                    // 0..31 (n offset)
  int c4  = (t & 7) * 4;                               // 0,4,..28 (d offset)

  const float* gsrc = x + ((size_t)b * NN + n0 + row) * DD + d0 + c4;
  unsigned      lds_off = (unsigned)(size_t)&tile[row][c4];
  unsigned long long gaddr = (unsigned long long)(size_t)gsrc;
  asm volatile("global_load_async_to_lds_b128 %0, %1, off"
               :: "v"(lds_off), "v"(gaddr) : "memory");
  asm volatile("s_wait_asynccnt 0x0" ::: "memory");
  __syncthreads();

  int dr = t >> 3;                                     // output d row 0..31
  int nc = (t & 7) * 4;                                // n col chunk
  __bf16* dstp = xT + ((size_t)b * DD + d0 + dr) * NN + n0 + nc;
#pragma unroll
  for (int i = 0; i < 4; ++i)
    dstp[i] = f2bf(tile[nc + i][dr]);
}

// ------- causal Toeplitz conv (WMMA) + residual*scale + LayerNorm1 -------
// grid (NN/16, NB), 256 threads = 8 waves; each block: 16 rows x full D.
__global__ void k_conv_ln(const float* __restrict__ x, const __bf16* __restrict__ xT,
                          const float* __restrict__ wconv,
                          const float* __restrict__ g, const float* __restrict__ bnv,
                          float* __restrict__ x1f, __bf16* __restrict__ x1h) {
  __shared__ float red[2][16];
  int tid = threadIdx.x;
  if (tid < 32) red[tid >> 4][tid & 15] = 0.0f;
  __syncthreads();

  int b = blockIdx.y;
  int n_base = blockIdx.x * 16;
  int lane = tid & 31, wid = tid >> 5;
  int col = lane & 15, kh = lane >> 4;
  int kh8 = kh * 8, kh16 = kh * 16;
  int nrow = n_base + col;                 // this lane's A row (M = lane&15)

  v8f zero = {0.f,0.f,0.f,0.f,0.f,0.f,0.f,0.f};
  v8f acc[8];
#pragma unroll
  for (int j = 0; j < 8; ++j) acc[j] = zero;

  const __bf16* xTb = xT + (size_t)b * DD * NN;
  int nk = (n_base + 16 + 31) >> 5;        // K tiles needed (triangular)

  for (int kt = 0; kt < nk; ++kt) {
    int k = kt * 32;
    // batch all 8 B fragments first (one load clause), build A while in flight
    v16bf bfr[8];
#pragma unroll
    for (int j = 0; j < 8; ++j) {
      int d = wid * 128 + j * 16 + col;    // B col (N = lane&15)
      bfr[j] = *(const v16bf*)(xTb + (size_t)d * NN + k + kh16);
    }
    // Toeplitz A fragment from w: A[n,m] = (n>=m) ? w[n-m] : 0
    // clamped index + select -> no EXEC-mask branching
    v16bf a;
#pragma unroll
    for (int i = 0; i < 8; ++i) {
      int i1 = nrow - (k + kh16 + i);
      int i2 = i1 - 8;
      float wa = wconv[i1 < 0 ? 0 : i1];
      float wb = wconv[i2 < 0 ? 0 : i2];
      a[i]     = f2bf(i1 < 0 ? 0.0f : wa);
      a[i + 8] = f2bf(i2 < 0 ? 0.0f : wb);
    }
#pragma unroll
    for (int j = 0; j < 8; ++j)
      acc[j] = __builtin_amdgcn_wmma_f32_16x16x32_bf16(false, a, false, bfr[j],
                                                       (short)0, acc[j], false, false);
  }

  // residual + per-position scale, accumulate row stats via LDS atomics
#pragma unroll
  for (int v = 0; v < 8; ++v) {
    int n = n_base + v + kh8;              // C row: M = v + 8*kh
    float sc = rsqrtf((float)(n + 1));
    float s_ = 0.f, q_ = 0.f;
#pragma unroll
    for (int j = 0; j < 8; ++j) {
      int d = wid * 128 + j * 16 + col;
      float xv = x[((size_t)b * NN + n) * DD + d];
      float z = xv + acc[j][v] * sc;
      acc[j][v] = z;
      s_ += z; q_ += z * z;
    }
    atomicAdd(&red[0][v + kh8], s_);
    atomicAdd(&red[1][v + kh8], q_);
  }
  __syncthreads();

  float gj[8], bj[8];
#pragma unroll
  for (int j = 0; j < 8; ++j) {
    int d = wid * 128 + j * 16 + col;
    gj[j] = g[d]; bj[j] = bnv[d];
  }
#pragma unroll
  for (int v = 0; v < 8; ++v) {
    int row = v + kh8;
    int n = n_base + row;
    float mu   = red[0][row] * (1.0f / DD);
    float var  = red[1][row] * (1.0f / DD) - mu * mu;
    float rstd = rsqrtf(var + LN_EPS);
#pragma unroll
    for (int j = 0; j < 8; ++j) {
      int d = wid * 128 + j * 16 + col;
      float o = (acc[j][v] - mu) * rstd * gj[j] + bj[j];
      size_t idx = ((size_t)b * NN + n) * DD + d;
      x1f[idx] = o;
      x1h[idx] = f2bf(o);
    }
  }
}

// ------------- expert GEMM1 + bias + exact GELU (WMMA) -------------
// wave strip: 16 tokens x 128 features; grid 2048 blocks x 256 threads
__global__ void k_gemm1_gelu(const __bf16* __restrict__ x1h, const __bf16* __restrict__ w1h,
                             const float* __restrict__ b1, __bf16* __restrict__ h, int e) {
  int lane = threadIdx.x & 31, wid = threadIdx.x >> 5;
  int col = lane & 15, kh8 = (lane >> 4) * 8, kh16 = (lane >> 4) * 16;
  int W = blockIdx.x * 8 + wid;
  int mt = W >> 5;                          // 0..511 token tiles
  int ns = W & 31;                          // 0..31 feature strips (128 wide)
  int lt = mt * 16 + col;
  size_t arow = ((size_t)(lt >> 11) * NN + (size_t)e * 2048 + (lt & 2047)) * DD;
  int f0 = ns * 128;

  v8f zero = {0.f,0.f,0.f,0.f,0.f,0.f,0.f,0.f};
  v8f acc[8];
#pragma unroll
  for (int j = 0; j < 8; ++j) acc[j] = zero;

  for (int kt = 0; kt < DD / 32; ++kt) {
    int k = kt * 32;
    // batch all loads (A + 8 B frags) -> one clause, one wait, 8 WMMAs
    v16bf bfr[8];
#pragma unroll
    for (int j = 0; j < 8; ++j)
      bfr[j] = *(const v16bf*)(w1h + (size_t)(f0 + j * 16 + col) * DD + k + kh16);
    v16bf a = load_a16(x1h + arow + k + kh16);
    if (kt + 1 < DD / 32) {
      __builtin_prefetch(x1h + arow + k + 32, 0, 0);
      __builtin_prefetch(w1h + (size_t)(f0 + col) * DD + k + 32 + kh16, 0, 0);
    }
#pragma unroll
    for (int j = 0; j < 8; ++j)
      acc[j] = __builtin_amdgcn_wmma_f32_16x16x32_bf16(false, a, false, bfr[j],
                                                       (short)0, acc[j], false, false);
  }
#pragma unroll
  for (int j = 0; j < 8; ++j) {
    int f = f0 + j * 16 + col;
    float bb = b1[f];
#pragma unroll
    for (int v = 0; v < 8; ++v) {
      int row = mt * 16 + v + kh8;
      h[(size_t)row * FF + f] = f2bf(gelu_exact(acc[j][v] + bb));
    }
  }
}

// ------------- expert GEMM2 + bias (WMMA), K = 4096 -------------
__global__ void k_gemm2(const __bf16* __restrict__ h, const __bf16* __restrict__ w2h,
                        const float* __restrict__ b2, float* __restrict__ y2) {
  int lane = threadIdx.x & 31, wid = threadIdx.x >> 5;
  int col = lane & 15, kh8 = (lane >> 4) * 8, kh16 = (lane >> 4) * 16;
  int W = blockIdx.x * 8 + wid;
  int mt = W >> 3;                          // 0..511
  int ds = W & 7;                           // 0..7 (1024/128)
  int d0 = ds * 128;
  size_t arow = (size_t)(mt * 16 + col) * FF;

  v8f zero = {0.f,0.f,0.f,0.f,0.f,0.f,0.f,0.f};
  v8f acc[8];
#pragma unroll
  for (int j = 0; j < 8; ++j) acc[j] = zero;

  for (int kt = 0; kt < FF / 32; ++kt) {
    int k = kt * 32;
    v16bf bfr[8];
#pragma unroll
    for (int j = 0; j < 8; ++j)
      bfr[j] = *(const v16bf*)(w2h + (size_t)(d0 + j * 16 + col) * FF + k + kh16);
    v16bf a = load_a16(h + arow + k + kh16);
    if (kt + 1 < FF / 32) {
      __builtin_prefetch(h + arow + k + 32, 0, 0);
      __builtin_prefetch(w2h + (size_t)(d0 + col) * FF + k + 32 + kh16, 0, 0);
    }
#pragma unroll
    for (int j = 0; j < 8; ++j)
      acc[j] = __builtin_amdgcn_wmma_f32_16x16x32_bf16(false, a, false, bfr[j],
                                                       (short)0, acc[j], false, false);
  }
#pragma unroll
  for (int j = 0; j < 8; ++j) {
    int d = d0 + j * 16 + col;
    float bb = b2[d];
#pragma unroll
    for (int v = 0; v < 8; ++v) {
      int row = mt * 16 + v + kh8;
      y2[(size_t)row * DD + d] = acc[j][v] + bb;
    }
  }
}

// ------------- final: out = LN(x1 + y2 * s) per token row -------------
__global__ void k_ln_out(const float* __restrict__ x1f,
                         const float* __restrict__ y20, const float* __restrict__ y21,
                         const float* __restrict__ s0, const float* __restrict__ s1,
                         const float* __restrict__ g0, const float* __restrict__ b0,
                         const float* __restrict__ g1, const float* __restrict__ b1,
                         float* __restrict__ out) {
  int gRow = blockIdx.x;                    // 0..NB*NN-1
  int b = gRow >> 12;
  int n = gRow & (NN - 1);
  int e = n >> 11;
  int lt = b * 2048 + (n & 2047);
  const float* y2 = e ? y21 : y20;
  float s = e ? s1[0] : s0[0];
  const float* g = e ? g1 : g0;
  const float* bb = e ? b1 : b0;
  const float* zx = x1f + (size_t)gRow * DD;
  const float* zy = y2  + (size_t)lt   * DD;

  int t = threadIdx.x;                      // 256 threads, 4 elems each
  float zv[4]; float ps = 0.f, pq = 0.f;
#pragma unroll
  for (int i = 0; i < 4; ++i) {
    float z = zx[t + i * 256] + zy[t + i * 256] * s;
    zv[i] = z; ps += z; pq += z * z;
  }
  __shared__ float ssum[256], ssq[256];
  ssum[t] = ps; ssq[t] = pq; __syncthreads();
  for (int off = 128; off > 0; off >>= 1) {
    if (t < off) { ssum[t] += ssum[t + off]; ssq[t] += ssq[t + off]; }
    __syncthreads();
  }
  float mu   = ssum[0] * (1.0f / DD);
  float var  = ssq[0]  * (1.0f / DD) - mu * mu;
  float rstd = rsqrtf(var + LN_EPS);
  float* o = out + (size_t)gRow * DD;
#pragma unroll
  for (int i = 0; i < 4; ++i) {
    int d = t + i * 256;
    o[d] = (zv[i] - mu) * rstd * g[d] + bb[d];
  }
}

extern "C" void kernel_launch(void* const* d_in, const int* in_sizes, int n_in,
                              void* d_out, int out_size, void* d_ws, size_t ws_size,
                              hipStream_t stream) {
  const float* x     = (const float*)d_in[0];
  const float* wconv = (const float*)d_in[1];
  const float* ng    = (const float*)d_in[2];
  const float* nb    = (const float*)d_in[3];
  const float* w1_0  = (const float*)d_in[4];
  const float* b1_0  = (const float*)d_in[5];
  const float* w2_0  = (const float*)d_in[6];
  const float* b2_0  = (const float*)d_in[7];
  const float* s_0   = (const float*)d_in[8];
  const float* lng_0 = (const float*)d_in[9];
  const float* lnb_0 = (const float*)d_in[10];
  const float* w1_1  = (const float*)d_in[11];
  const float* b1_1  = (const float*)d_in[12];
  const float* w2_1  = (const float*)d_in[13];
  const float* b2_1  = (const float*)d_in[14];
  const float* s_1   = (const float*)d_in[15];
  const float* lng_1 = (const float*)d_in[16];
  const float* lnb_1 = (const float*)d_in[17];

  char* ws = (char*)d_ws;
  const size_t MB = 1ull << 20;
  float*  x1f  = (float*) (ws + 0);            // 64 MB
  __bf16* x1h  = (__bf16*)(ws + 64  * MB);     // 32 MB
  __bf16* xT   = (__bf16*)(ws + 96  * MB);     // 32 MB
  __bf16* w1h0 = (__bf16*)(ws + 128 * MB);     // 8 MB each
  __bf16* w2h0 = (__bf16*)(ws + 136 * MB);
  __bf16* w1h1 = (__bf16*)(ws + 144 * MB);
  __bf16* w2h1 = (__bf16*)(ws + 152 * MB);
  __bf16* hbuf = (__bf16*)(ws + 160 * MB);     // 64 MB (reused per expert)
  float*  y20  = (float*) (ws + 224 * MB);     // 32 MB
  float*  y21  = (float*) (ws + 256 * MB);     // 32 MB

  const int WEL = FF * DD;                      // 4M elements per weight matrix
  k_cvt_bf16<<<(WEL + 255) / 256, 256, 0, stream>>>(w1_0, w1h0, WEL);
  k_cvt_bf16<<<(WEL + 255) / 256, 256, 0, stream>>>(w2_0, w2h0, WEL);
  k_cvt_bf16<<<(WEL + 255) / 256, 256, 0, stream>>>(w1_1, w1h1, WEL);
  k_cvt_bf16<<<(WEL + 255) / 256, 256, 0, stream>>>(w2_1, w2h1, WEL);

  k_transpose<<<dim3(NN / 32, DD / 32, NB), 256, 0, stream>>>(x, xT);

  k_conv_ln<<<dim3(NN / 16, NB), 256, 0, stream>>>(x, xT, wconv, ng, nb, x1f, x1h);

  // expert 0
  k_gemm1_gelu<<<2048, 256, 0, stream>>>(x1h, w1h0, b1_0, hbuf, 0);
  k_gemm2<<<512, 256, 0, stream>>>(hbuf, w2h0, b2_0, y20);
  // expert 1
  k_gemm1_gelu<<<2048, 256, 0, stream>>>(x1h, w1h1, b1_1, hbuf, 1);
  k_gemm2<<<512, 256, 0, stream>>>(hbuf, w2h1, b2_1, y21);

  k_ln_out<<<NB * NN, 256, 0, stream>>>(x1f, y20, y21, s_0, s_1,
                                        lng_0, lnb_0, lng_1, lnb_1, (float*)d_out);
}
